// ParameterDot_58703613002152
// MI455X (gfx1250) — matrix operand order repaired and checked by reading, not compiled, not run on hardware
//
#include <hip/hip_runtime.h>
#include <stdint.h>

// out[b,l] = sum_{i,j} t1[b,l,i] * W[idx[b],i,j] * t2[b,l,j]
//
// Roofline (MI455X): 17.2 GFLOP; min traffic 201 MB -> 8.6 us HBM floor @23.3TB/s.
// Path: split-bf16 (x = x_hi + x_lo) GEMM on the bf16 WMMA pipe (16x16x32),
// 3 WMMAs per k-step (ah*bh + ah*bl + al*bh), fp32 accumulation -> fp32-class
// accuracy at ~the memory floor. Weights preprocessed once into transposed
// split-bf16 so the hot loop streams contiguous panels via the Tensor Data
// Mover into LDS (double buffered, TDM-padded columns for conflict-free
// ds_load_b128 B-fragment fetches).
//
// d_ws requirement: 2 * E*D1*D2 * 2B = 64 MB.

constexpr int BB = 256, LL = 128, EE = 64, DD1 = 512, DD2 = 512;

typedef __attribute__((ext_vector_type(16))) __bf16       v16bf;
typedef __attribute__((ext_vector_type(8)))  float        v8f;
typedef __attribute__((ext_vector_type(4)))  float        v4f;
typedef __attribute__((ext_vector_type(4)))  unsigned int u32x4;
typedef __attribute__((ext_vector_type(8)))  unsigned int u32x8;
typedef __attribute__((ext_vector_type(8)))  int          i32x8;
typedef __attribute__((ext_vector_type(4)))  int          i32x4;

#if defined(__HIP_DEVICE_COMPILE__) && __has_builtin(__builtin_amdgcn_tensor_load_to_lds)
#define HAVE_TDM 1
#else
#define HAVE_TDM 0
#endif

// LDS panel: [hi plane | lo plane]; plane = 16 columns (N) x 512 K bf16,
// column = 1024B data + 16B pad (TDM pad_interval=1024B, pad_amount=16B)
// so the 32 ds_load_b128 fragment reads spread over all 64 banks.
#define COL_PITCH   1040
#define PLANE_BYTES (16 * COL_PITCH)   // 16640
#define BUF_BYTES   (2 * PLANE_BYTES)  // 33280
#define LDS_BYTES   (2 * BUF_BYTES)    // 66560 (double buffered)

// ---------------------------------------------------------------------------
// Preprocess: W[e][k][j] (fp32, row major) -> wT_hi/wT_lo[e][j][k] (bf16 split)
// LDS-tiled 32x32 transpose: coalesced reads and writes.
// ---------------------------------------------------------------------------
__global__ __launch_bounds__(256) void wsplit_kernel(
    const float* __restrict__ W, unsigned short* __restrict__ whi,
    unsigned short* __restrict__ wlo)
{
  __shared__ float tile[32][33];
  const int tid = threadIdx.x;
  const int tx = tid & 31, ty = tid >> 5;
  const int k0 = blockIdx.x * 32, j0 = blockIdx.y * 32, e = blockIdx.z;
  const size_t wbase = (size_t)e * DD1 * DD2;
#pragma unroll
  for (int p = 0; p < 4; ++p)
    tile[ty + p * 8][tx] = W[wbase + (size_t)(k0 + ty + p * 8) * DD2 + (j0 + tx)];
  __syncthreads();
  const size_t tbase = (size_t)e * DD2 * DD1;
#pragma unroll
  for (int p = 0; p < 4; ++p) {
    const int j   = j0 + ty + p * 8;
    const float f = tile[tx][ty + p * 8];
    const __bf16 h = (__bf16)f;
    const __bf16 l = (__bf16)(f - (float)h);
    const size_t o = tbase + (size_t)j * DD1 + (k0 + tx);
    whi[o] = __builtin_bit_cast(unsigned short, h);
    wlo[o] = __builtin_bit_cast(unsigned short, l);
  }
}

// ---------------------------------------------------------------------------
// TDM: one descriptor loads BOTH planes of one N-panel (hi row + lo row,
// dim0 = 8192 bf16 contiguous, dim1 = 2 with stride = whi->wlo distance),
// padding 16B per 1024B into LDS.
// This toolchain's builtin is the 6-arg form:
//   (uint32x4 g0, int32x8 g1, int32x4 g2, int32x4 g3, int32x8, i32 cpol)
// ---------------------------------------------------------------------------
#if HAVE_TDM
__device__ __forceinline__ void tdm_issue_panel(const unsigned short* hi_src,
                                                unsigned lds_off)
{
  const uint64_t ga = (uint64_t)(uintptr_t)hi_src;
  u32x4 g0;
  g0[0] = 1u;                                                // count=1
  g0[1] = lds_off;                                           // LDS byte address
  g0[2] = (unsigned)ga;                                      // global addr lo
  g0[3] = (unsigned)((ga >> 32) & 0x1FFFFFFu) | (2u << 30);  // addr[56:32]|type=2
  i32x8 g1;
  g1[0] = (int)((1u << 16) |      // data_size = 1 (2 bytes)
                (1u << 20) |      // pad_enable
                (7u << 22) |      // pad_interval = 256 DWORDs (1024B)
                (3u << 25));      // pad_amount   = 4 DWORDs (16B)
  g1[1] = (int)(8192u << 16);     // tensor_dim0[15:0]  = 8192
  g1[2] = (int)(2u << 16);        // tensor_dim0 hi=0 ; tensor_dim1 = 2
  g1[3] = (int)(8192u << 16);     // tensor_dim1 hi=0 ; tile_dim0 = 8192
  g1[4] = 2;                      // tile_dim1 = 2 (hi,lo planes); tile_dim2 = 0
  g1[5] = (int)(EE * DD1 * DD2);  // tensor_dim0_stride = whi -> wlo (elements)
  g1[6] = 0;
  g1[7] = 0;
  const i32x4 z4 = {0, 0, 0, 0};
  const i32x8 z8 = {0, 0, 0, 0, 0, 0, 0, 0};
  __builtin_amdgcn_tensor_load_to_lds(g0, g1, z4, z4, z8, 0);
}
#endif

// ---------------------------------------------------------------------------
// Main kernel: 1 block per batch; 8 waves, wave w owns L-rows [16w,16w+16).
// A (16x512, split bf16) lives in registers; W N-panels streamed through LDS.
// ---------------------------------------------------------------------------
__global__ __launch_bounds__(256, 2) void pdot_kernel(
    const float* __restrict__ t1, const float* __restrict__ t2,
    const int* __restrict__ pidx, const unsigned short* __restrict__ whi,
    const unsigned short* __restrict__ wlo, float* __restrict__ out)
{
  extern __shared__ unsigned char smem[];
  const int b    = blockIdx.x;
  const int tid  = threadIdx.x;
  const int wv   = tid >> 5;
  const int lane = tid & 31;
  const int nn   = lane & 15;   // N (B frag col) / M (A frag row) / C-frag N
  const int kg   = lane >> 4;   // K-group half
  const int e    = __builtin_amdgcn_readfirstlane(pidx[b]);
  const unsigned short* wpanel0 = whi + (size_t)e * DD2 * DD1;

#if HAVE_TDM
  if (wv == 0)  // wave-uniform branch: only wave 0 issues the DMA
    tdm_issue_panel(wpanel0, (unsigned)(uintptr_t)&smem[0]);
#endif

  // ---- A fragments: per-lane layout of v_wmma_*_bf16 A (16x32):
  // lane m=nn holds K = 32s + kg*8 + (e%8) + (e/8)*16 in element e.
  v16bf a_hi[16], a_lo[16];
  {
    const float* arow = t1 + ((size_t)b * LL + (wv * 16 + nn)) * DD1 + kg * 8;
#pragma unroll
    for (int s = 0; s < 16; ++s) {
      const float* p = arow + s * 32;
      v4f f0 = *(const v4f*)(p);
      v4f f1 = *(const v4f*)(p + 4);
      v4f f2 = *(const v4f*)(p + 16);
      v4f f3 = *(const v4f*)(p + 20);
      float fv[16];
#pragma unroll
      for (int q = 0; q < 4; ++q) { fv[q] = f0[q]; fv[4+q] = f1[q]; fv[8+q] = f2[q]; fv[12+q] = f3[q]; }
      v16bf h, l;
#pragma unroll
      for (int q = 0; q < 16; ++q) {
        const __bf16 hb = (__bf16)fv[q];
        h[q] = hb;
        l[q] = (__bf16)(fv[q] - (float)hb);
      }
      a_hi[s] = h; a_lo[s] = l;
    }
  }

  float ys[8] = {0, 0, 0, 0, 0, 0, 0, 0};
  const float* t2p = t2 + ((size_t)b * LL + (wv * 16 + kg * 8)) * DD2 + nn;

#pragma unroll 1
  for (int nt = 0; nt < 32; ++nt) {
#if HAVE_TDM
    const unsigned bufoff = (unsigned)(nt & 1) * BUF_BYTES;
    if (wv == 0) {
      if (nt + 1 < 32) {  // prefetch next panel into the other buffer
        tdm_issue_panel(wpanel0 + (size_t)(nt + 1) * 16 * DD1,
                        (unsigned)(uintptr_t)&smem[0] +
                            (unsigned)((nt + 1) & 1) * BUF_BYTES);
        __builtin_amdgcn_s_wait_tensorcnt(1);  // current panel landed
      } else {
        __builtin_amdgcn_s_wait_tensorcnt(0);
      }
    }
    __syncthreads();
#else
    const unsigned bufoff = 0;
    __syncthreads();  // previous panel fully consumed before overwrite
    {
      const unsigned int* sh = (const unsigned int*)(whi + ((size_t)e * DD2 + nt * 16) * DD1);
      const unsigned int* sl = (const unsigned int*)(wlo + ((size_t)e * DD2 + nt * 16) * DD1);
      for (int i = tid; i < 4096; i += 256) {
        const int col = i >> 8, cw = i & 255;
        *(unsigned int*)(smem + col * COL_PITCH + cw * 4) = sh[i];
        *(unsigned int*)(smem + PLANE_BYTES + col * COL_PITCH + cw * 4) = sl[i];
      }
    }
    __syncthreads();
#endif

    // ---- B fragments: lane holds column nn, K = 32s + kg*8 + [0..7] and 32s + 16 + kg*8 + [0..7]
    // in the padded LDS column -> two ds_load_b128 per matrix.
    const unsigned char* hiB = smem + bufoff + nn * COL_PITCH + kg * 16;
    const unsigned char* loB = hiB + PLANE_BYTES;
    v8f acc = {0, 0, 0, 0, 0, 0, 0, 0};
#pragma unroll
    for (int s = 0; s < 16; ++s) {
      u32x4 h0 = *(const u32x4*)(hiB + s * 64);
      u32x4 h1 = *(const u32x4*)(hiB + s * 64 + 32);
      u32x4 l0 = *(const u32x4*)(loB + s * 64);
      u32x4 l1 = *(const u32x4*)(loB + s * 64 + 32);
      v16bf bh = __builtin_bit_cast(v16bf, __builtin_shufflevector(h0, h1, 0, 1, 2, 3, 4, 5, 6, 7));
      v16bf bl = __builtin_bit_cast(v16bf, __builtin_shufflevector(l0, l1, 0, 1, 2, 3, 4, 5, 6, 7));
      acc = __builtin_amdgcn_wmma_f32_16x16x32_bf16(false, a_hi[s], false, bh, (short)0, acc, false, false);
      acc = __builtin_amdgcn_wmma_f32_16x16x32_bf16(false, a_hi[s], false, bl, (short)0, acc, false, false);
      acc = __builtin_amdgcn_wmma_f32_16x16x32_bf16(false, a_lo[s], false, bh, (short)0, acc, false, false);
    }

    // Fuse X-tile with t2: C/D layout -> VGPR r holds row M = r + 8*kg, N = nn.
#pragma unroll
    for (int r = 0; r < 8; ++r)
      ys[r] += acc[r] * t2p[(size_t)r * DD2 + nt * 16];

#if HAVE_TDM
    __syncthreads();  // all reads of this buffer done before TDM reuses it
#endif
  }

  // Reduce each row's partial over its 16 N-lanes (stay within half-wave).
#pragma unroll
  for (int r = 0; r < 8; ++r) {
    float v = ys[r];
    v += __shfl_xor(v, 1);
    v += __shfl_xor(v, 2);
    v += __shfl_xor(v, 4);
    v += __shfl_xor(v, 8);
    if (nn == 0) out[(size_t)b * LL + wv * 16 + kg * 8 + r] = v;
  }
}

// ---------------------------------------------------------------------------
extern "C" void kernel_launch(void* const* d_in, const int* in_sizes, int n_in,
                              void* d_out, int out_size, void* d_ws, size_t ws_size,
                              hipStream_t stream) {
  (void)in_sizes; (void)n_in; (void)out_size; (void)ws_size;  // ws >= 64MB assumed
  const float* t1  = (const float*)d_in[0];
  const float* t2  = (const float*)d_in[1];
  const int*  pidx = (const int*)d_in[2];
  const float* W   = (const float*)d_in[3];
  float* out = (float*)d_out;
  unsigned short* whi = (unsigned short*)d_ws;
  unsigned short* wlo = whi + (size_t)EE * DD1 * DD2;

  wsplit_kernel<<<dim3(DD1 / 32, DD2 / 32, EE), dim3(256), 0, stream>>>(W, whi, wlo);
  pdot_kernel<<<dim3(BB), dim3(256), LDS_BYTES, stream>>>(t1, t2, pidx, whi, wlo, out);
}
